// Linear4Bit_30906584662422
// MI455X (gfx1250) — matrix operand order repaired and checked by reading, not compiled, not run on hardware
//
#include <hip/hip_runtime.h>

typedef __attribute__((ext_vector_type(16))) _Float16 v16h;
typedef __attribute__((ext_vector_type(8)))  _Float16 v8h;
typedef __attribute__((ext_vector_type(4)))  _Float16 v4h;
typedef __attribute__((ext_vector_type(8)))  float    v8f;
typedef __attribute__((ext_vector_type(4)))  float    v4f;
typedef __attribute__((ext_vector_type(4)))  int      v4i;

#define K_IN   4096
#define N_OUT  4096
#define M_TOT  8192
#define BM     128
#define BN     128
#define BK     64
#define LDK    (BK + 8)      // padded halfs per LDS row (144 B stride)

union Frag16 { v16h v; v8h h[2]; };

// ---------------------------------------------------------------------------
// Kernel 1: dequantize 4-bit weights -> f16 (row-major [N_OUT][K_IN])
// Each thread handles one int4 = 4 packed bytes = 8 weights.
// Flat half index: group g covers halves [g*256, g*256+256).
// ---------------------------------------------------------------------------
__global__ __launch_bounds__(256)
void dequant_w_kernel(const int* __restrict__ wq,
                      const float* __restrict__ wscale,
                      _Float16* __restrict__ wf) {
    const int idx = blockIdx.x * 256 + threadIdx.x;       // 0 .. 2097151
    const v4i b = *(const v4i*)(wq + (size_t)idx * 4);
    const int g = idx >> 5;                               // 32 int4 per group
    const float s = wscale[g];
    const float a = s * (2.0f / 15.0f);
    v8h h;
    h[0] = (_Float16)(a * (float)( b.x       & 15) - s);
    h[1] = (_Float16)(a * (float)((b.x >> 4) & 15) - s);
    h[2] = (_Float16)(a * (float)( b.y       & 15) - s);
    h[3] = (_Float16)(a * (float)((b.y >> 4) & 15) - s);
    h[4] = (_Float16)(a * (float)( b.z       & 15) - s);
    h[5] = (_Float16)(a * (float)((b.z >> 4) & 15) - s);
    h[6] = (_Float16)(a * (float)( b.w       & 15) - s);
    h[7] = (_Float16)(a * (float)((b.w >> 4) & 15) - s);
    *(v8h*)(wf + (size_t)g * 256 + (idx & 31) * 8) = h;
}

// ---------------------------------------------------------------------------
// Kernel 2: convert x fp32 -> f16. Each thread handles 8 floats.
// ---------------------------------------------------------------------------
__global__ __launch_bounds__(256)
void cvt_x_kernel(const float* __restrict__ x, _Float16* __restrict__ xh) {
    const size_t idx = (size_t)(blockIdx.x * 256 + threadIdx.x) * 8;
    const v4f a = *(const v4f*)(x + idx);
    const v4f b = *(const v4f*)(x + idx + 4);
    v8h h;
    h[0] = (_Float16)a.x; h[1] = (_Float16)a.y;
    h[2] = (_Float16)a.z; h[3] = (_Float16)a.w;
    h[4] = (_Float16)b.x; h[5] = (_Float16)b.y;
    h[6] = (_Float16)b.z; h[7] = (_Float16)b.w;
    *(v8h*)(xh + idx) = h;
}

// ---------------------------------------------------------------------------
// Kernel 3: pure-f16 WMMA GEMM. out[m,n] = sum_k xh[m,k]*wf[n,k] + bias[n]
// Block 256 threads = 8 waves; tile 128x128, K-tile 64; wave = 32x64.
// ---------------------------------------------------------------------------
__global__ __launch_bounds__(256)
void gemm_f16_kernel(const _Float16* __restrict__ xh,
                     const _Float16* __restrict__ wf,
                     const float* __restrict__ bias,
                     float* __restrict__ out) {
    __shared__ _Float16 sX[BM * LDK];
    __shared__ _Float16 sW[BN * LDK];

    const int tid     = threadIdx.x;
    const int lane    = tid & 31;
    const int wave    = tid >> 5;
    const int lanelo  = lane & 15;
    const int halfSel = lane >> 4;

    const int m0 = blockIdx.y * BM;
    const int n0 = blockIdx.x * BN;
    const int wm = (wave & 3) * 32;
    const int wn = (wave >> 2) * 64;

    v8f acc[2][4];
#pragma unroll
    for (int mi = 0; mi < 2; ++mi)
#pragma unroll
        for (int ni = 0; ni < 4; ++ni)
#pragma unroll
            for (int e = 0; e < 8; ++e) acc[mi][ni][e] = 0.0f;

    for (int kt = 0; kt < K_IN / BK; ++kt) {
        const int k0 = kt * BK;

        // global f16 loads into registers: 128 rows x 8 chunks of 8 halfs, 4/thread each
        v8h xr[4], wr[4];
#pragma unroll
        for (int r = 0; r < 4; ++r) {
            const int idx = r * 256 + tid;
            const int row = idx >> 3;
            const int c8  = idx & 7;
            xr[r] = *(const v8h*)(xh + (size_t)(m0 + row) * K_IN + k0 + c8 * 8);
            wr[r] = *(const v8h*)(wf + (size_t)(n0 + row) * K_IN + k0 + c8 * 8);
        }

        __syncthreads();

#pragma unroll
        for (int r = 0; r < 4; ++r) {
            const int idx = r * 256 + tid;
            const int row = idx >> 3;
            const int c8  = idx & 7;
            *(v8h*)(sX + row * LDK + c8 * 8) = xr[r];
            *(v8h*)(sW + row * LDK + c8 * 8) = wr[r];
        }

        __syncthreads();

#pragma unroll
        for (int ks = 0; ks < 2; ++ks) {
            const int kb = ks * 32;
            Frag16 afrag[2];
#pragma unroll
            for (int mi = 0; mi < 2; ++mi) {
                const _Float16* p = sX + (wm + mi * 16 + lanelo) * LDK + kb + halfSel * 8;
                afrag[mi].h[0] = *(const v8h*)(p);        // K +0..7
                afrag[mi].h[1] = *(const v8h*)(p + 16);   // K +16..23
            }
            Frag16 bfrag[4];
#pragma unroll
            for (int ni = 0; ni < 4; ++ni) {
                const _Float16* p = sW + (wn + ni * 16 + lanelo) * LDK + kb + halfSel * 8;
                bfrag[ni].h[0] = *(const v8h*)(p);
                bfrag[ni].h[1] = *(const v8h*)(p + 16);
            }
#pragma unroll
            for (int mi = 0; mi < 2; ++mi)
#pragma unroll
                for (int ni = 0; ni < 4; ++ni)
                    acc[mi][ni] = __builtin_amdgcn_wmma_f32_16x16x32_f16(
                        false, afrag[mi].v, false, bfrag[ni].v,
                        (short)0, acc[mi][ni], false, false);
        }
    }

#pragma unroll
    for (int mi = 0; mi < 2; ++mi) {
#pragma unroll
        for (int ni = 0; ni < 4; ++ni) {
            const int n     = n0 + wn + ni * 16 + lanelo;
            const int mbase = m0 + wm + mi * 16 + halfSel * 8;
            const float b   = bias[n];
#pragma unroll
            for (int r = 0; r < 8; ++r) {
                out[(size_t)(mbase + r) * N_OUT + n] = acc[mi][ni][r] + b;
            }
        }
    }
}

// ---------------------------------------------------------------------------
// Fallback: fused kernel (dequant + convert inside the loop) if ws too small.
// ---------------------------------------------------------------------------
__global__ __launch_bounds__(256)
void l4b_fused_kernel(const float* __restrict__ x,
                      const int*   __restrict__ wq,
                      const float* __restrict__ wscale,
                      const float* __restrict__ bias,
                      float* __restrict__ out) {
    __shared__ _Float16 sX[BM * LDK];
    __shared__ _Float16 sW[BN * LDK];

    const int tid     = threadIdx.x;
    const int lane    = tid & 31;
    const int wave    = tid >> 5;
    const int lanelo  = lane & 15;
    const int halfSel = lane >> 4;
    const int m0 = blockIdx.y * BM;
    const int n0 = blockIdx.x * BN;
    const int wm = (wave & 3) * 32;
    const int wn = (wave >> 2) * 64;

    v8f acc[2][4];
#pragma unroll
    for (int mi = 0; mi < 2; ++mi)
#pragma unroll
        for (int ni = 0; ni < 4; ++ni)
#pragma unroll
            for (int e = 0; e < 8; ++e) acc[mi][ni][e] = 0.0f;

    for (int kt = 0; kt < K_IN / BK; ++kt) {
        const int k0 = kt * BK;
        v4f xr[8];
#pragma unroll
        for (int r = 0; r < 8; ++r) {
            const int idx = r * 256 + tid;
            const int row = idx >> 4;
            const int c4  = idx & 15;
            xr[r] = *(const v4f*)(x + (size_t)(m0 + row) * K_IN + k0 + c4 * 4);
        }
        v4i   wr[4];
        float sc[4];
#pragma unroll
        for (int r = 0; r < 4; ++r) {
            const int idx = r * 256 + tid;
            const int row = idx >> 3;
            const int q   = idx & 7;
            const int g   = (n0 + row) * (K_IN / 256) + (k0 >> 8);
            wr[r] = *(const v4i*)(wq + (size_t)g * 128 + ((k0 & 255) >> 1) + q * 4);
            sc[r] = wscale[g];
        }

        __syncthreads();

#pragma unroll
        for (int r = 0; r < 8; ++r) {
            const int idx = r * 256 + tid;
            const int row = idx >> 4;
            const int c4  = idx & 15;
            v4h h;
            h[0] = (_Float16)xr[r].x; h[1] = (_Float16)xr[r].y;
            h[2] = (_Float16)xr[r].z; h[3] = (_Float16)xr[r].w;
            *(v4h*)(sX + row * LDK + c4 * 4) = h;
        }
#pragma unroll
        for (int r = 0; r < 4; ++r) {
            const int idx = r * 256 + tid;
            const int row = idx >> 3;
            const int q   = idx & 7;
            const float s = sc[r];
            const float a = s * (2.0f / 15.0f);
            const int w0 = wr[r].x, w1 = wr[r].y, w2 = wr[r].z, w3 = wr[r].w;
            v8h h;
            h[0] = (_Float16)(a * (float)( w0       & 15) - s);
            h[1] = (_Float16)(a * (float)((w0 >> 4) & 15) - s);
            h[2] = (_Float16)(a * (float)( w1       & 15) - s);
            h[3] = (_Float16)(a * (float)((w1 >> 4) & 15) - s);
            h[4] = (_Float16)(a * (float)( w2       & 15) - s);
            h[5] = (_Float16)(a * (float)((w2 >> 4) & 15) - s);
            h[6] = (_Float16)(a * (float)( w3       & 15) - s);
            h[7] = (_Float16)(a * (float)((w3 >> 4) & 15) - s);
            *(v8h*)(sW + row * LDK + q * 8) = h;
        }

        __syncthreads();

#pragma unroll
        for (int ks = 0; ks < 2; ++ks) {
            const int kb = ks * 32;
            Frag16 afrag[2];
#pragma unroll
            for (int mi = 0; mi < 2; ++mi) {
                const _Float16* p = sX + (wm + mi * 16 + lanelo) * LDK + kb + halfSel * 8;
                afrag[mi].h[0] = *(const v8h*)(p);
                afrag[mi].h[1] = *(const v8h*)(p + 16);
            }
            Frag16 bfrag[4];
#pragma unroll
            for (int ni = 0; ni < 4; ++ni) {
                const _Float16* p = sW + (wn + ni * 16 + lanelo) * LDK + kb + halfSel * 8;
                bfrag[ni].h[0] = *(const v8h*)(p);
                bfrag[ni].h[1] = *(const v8h*)(p + 16);
            }
#pragma unroll
            for (int mi = 0; mi < 2; ++mi)
#pragma unroll
                for (int ni = 0; ni < 4; ++ni)
                    acc[mi][ni] = __builtin_amdgcn_wmma_f32_16x16x32_f16(
                        false, afrag[mi].v, false, bfrag[ni].v,
                        (short)0, acc[mi][ni], false, false);
        }
    }

#pragma unroll
    for (int mi = 0; mi < 2; ++mi) {
#pragma unroll
        for (int ni = 0; ni < 4; ++ni) {
            const int n     = n0 + wn + ni * 16 + lanelo;
            const int mbase = m0 + wm + mi * 16 + halfSel * 8;
            const float b   = bias[n];
#pragma unroll
            for (int r = 0; r < 8; ++r) {
                out[(size_t)(mbase + r) * N_OUT + n] = acc[mi][ni][r] + b;
            }
        }
    }
}

extern "C" void kernel_launch(void* const* d_in, const int* in_sizes, int n_in,
                              void* d_out, int out_size, void* d_ws, size_t ws_size,
                              hipStream_t stream) {
    const float* x      = (const float*)d_in[0];   // (4,2048,4096) fp32
    const int*   wq     = (const int*)d_in[1];     // (65536,128) int32 (byte values)
    const float* wscale = (const float*)d_in[2];   // (65536,1) fp32
    const float* bias   = (const float*)d_in[3];   // (4096,) fp32
    float* out = (float*)d_out;                    // (4,2048,4096) fp32

    const size_t wf_bytes = (size_t)N_OUT * K_IN * sizeof(_Float16);  // 33.5 MB
    const size_t xh_bytes = (size_t)M_TOT * K_IN * sizeof(_Float16);  // 67 MB

    dim3 grid(N_OUT / BN, M_TOT / BM, 1);          // 32 x 64
    dim3 block(256, 1, 1);

    if (ws_size >= wf_bytes + xh_bytes) {
        _Float16* wf = (_Float16*)d_ws;
        _Float16* xh = (_Float16*)((char*)d_ws + wf_bytes);
        // 65536*128/4 int4s = 2,097,152 threads
        dequant_w_kernel<<<dim3(8192), block, 0, stream>>>(wq, wscale, wf);
        // 33,554,432 floats / 8 per thread = 4,194,304 threads
        cvt_x_kernel<<<dim3(16384), block, 0, stream>>>(x, xh);
        gemm_f16_kernel<<<grid, block, 0, stream>>>(xh, wf, bias, out);
    } else {
        l4b_fused_kernel<<<grid, block, 0, stream>>>(x, wq, wscale, bias, out);
    }
}
